// MambaScratchBlock_10213432230008
// MI455X (gfx1250) — matrix operand order repaired and checked by reading, not compiled, not run on hardware
//
#include <hip/hip_runtime.h>
#include <hip/hip_bf16.h>
#include <math.h>

// ---------------- problem constants ----------------
#define B_    2
#define L_    2048
#define D_    768
#define DI_   1536      // D_INNER
#define NS_   16        // D_STATE
#define DTR_  48        // DT_RANK
#define ROWS_ (B_ * L_) // 4096 tokens

typedef _Float16 v16h __attribute__((ext_vector_type(16)));
typedef _Float16 v8h  __attribute__((ext_vector_type(8)));
typedef float    v8f  __attribute__((ext_vector_type(8)));
typedef int      v4i_ __attribute__((vector_size(16)));   // matches builtin param type

// =====================================================================
// CDNA5 async global->LDS copy (ASYNCcnt path) with safe fallback.
// Builtin signature (from clang diagnostic):
//   (v4i __device__*, v4i __shared__*, imm offset, imm cpol)
// =====================================================================
__device__ __forceinline__ void async_copy_b128(void* dst_lds, const void* src_glob)
{
#if __has_builtin(__builtin_amdgcn_global_load_async_to_lds_b128)
    __builtin_amdgcn_global_load_async_to_lds_b128(
        (__attribute__((address_space(1))) v4i_*)src_glob,
        (__attribute__((address_space(3))) v4i_*)dst_lds,
        0, 0);
#else
    *(v8h*)dst_lds = *(const v8h*)src_glob;
#endif
}

__device__ __forceinline__ void wait_async_lds()
{
#if __has_builtin(__builtin_amdgcn_global_load_async_to_lds_b128)
#if __has_builtin(__builtin_amdgcn_s_wait_asynccnt)
    __builtin_amdgcn_s_wait_asynccnt(0);
#else
    asm volatile("s_wait_asynccnt 0x0" ::: "memory");
#endif
#endif
}

// =====================================================================
// LayerNorm over D, emit f16 for WMMA GEMM.  One block per token.
// =====================================================================
__global__ __launch_bounds__(256)
void mamba_layernorm_f16(const float* __restrict__ x,
                         const float* __restrict__ w,
                         const float* __restrict__ b,
                         _Float16* __restrict__ out)
{
    const int tok = blockIdx.x;
    const float* xr = x + (size_t)tok * D_;
    __shared__ float red0[8];
    __shared__ float red1[8];

    float s = 0.f, s2 = 0.f;
    for (int i = threadIdx.x; i < D_; i += 256) {
        float v = xr[i];
        s += v; s2 += v * v;
    }
    for (int off = 16; off > 0; off >>= 1) {
        s  += __shfl_down(s,  off, 32);
        s2 += __shfl_down(s2, off, 32);
    }
    const int wave = threadIdx.x >> 5, lane = threadIdx.x & 31;
    if (lane == 0) { red0[wave] = s; red1[wave] = s2; }
    __syncthreads();
    if (threadIdx.x == 0) {
        float a = 0.f, c = 0.f;
        #pragma unroll
        for (int i = 0; i < 8; ++i) { a += red0[i]; c += red1[i]; }
        red0[0] = a; red1[0] = c;
    }
    __syncthreads();
    const float mean = red0[0] * (1.0f / D_);
    const float var  = red1[0] * (1.0f / D_) - mean * mean;
    const float rstd = rsqrtf(var + 1e-5f);

    _Float16* orow = out + (size_t)tok * D_;
    for (int i = threadIdx.x; i < D_; i += 256)
        orow[i] = (_Float16)((xr[i] - mean) * rstd * w[i] + b[i]);
}

// =====================================================================
// Convert fp32 weight (K x N, row major) -> f16 transposed (Npad x Kpad),
// zero padded.
// =====================================================================
__global__ __launch_bounds__(256)
void mamba_convert_transpose(const float* __restrict__ W, _Float16* __restrict__ WT,
                             int K, int N, int Kpad, int Npad)
{
    int idx = blockIdx.x * 256 + threadIdx.x;
    if (idx >= Kpad * Npad) return;
    int n = idx / Kpad;
    int k = idx - n * Kpad;
    float v = (k < K && n < N) ? W[(size_t)k * N + n] : 0.0f;
    WT[idx] = (_Float16)v;
}

// =====================================================================
// WMMA GEMM:  C[M x N] = A[M x K] * BT[N x K]^T + bias, optional softplus.
//   - A, BT f16 row-major; M % 128 == 0, N % 128 == 0, K % 64 == 0.
//   - block = 256 threads (8 waves); block tile 128 x 128.
//     Wave tile 32 x 64 (waves arranged 4 rows x 2 cols):
//     2x4 = 8 f32 16x16 accumulators, B fragments reused across 2 A frags.
//   - K staged 64-wide into double-buffered LDS via async global->LDS DMA;
//     next stage issued before current stage's 16 WMMAs (latency hiding).
//   epilogue: 0 = bias, 1 = softplus(bias-added) + 1e-4
// =====================================================================
__global__ __launch_bounds__(256)
void mamba_gemm_wmma(const _Float16* __restrict__ A,
                     const _Float16* __restrict__ BT,
                     const float*    __restrict__ bias,
                     float*          __restrict__ C,
                     int M, int N, int K, int NbiasReal, int epilogue)
{
    __shared__ alignas(32) _Float16 lds_a[2][128 * 64];
    __shared__ alignas(32) _Float16 lds_b[2][128 * 64];

    const int tid  = threadIdx.x;
    const int wave = tid >> 5;
    const int lane = tid & 31;
    const int wr   = wave >> 1;     // 0..3 : 32-row group
    const int wc   = wave & 1;      // 0..1 : 64-col group
    const int bm   = blockIdx.y * 128;
    const int bn   = blockIdx.x * 128;

    const int mrow = lane & 15;     // M row in 16-subtile / N col in 16-subtile
    const int h    = lane >> 4;     // K-half selector

    v8f acc[2][4];
    #pragma unroll
    for (int am = 0; am < 2; ++am)
        #pragma unroll
        for (int tn = 0; tn < 4; ++tn)
            #pragma unroll
            for (int r = 0; r < 8; ++r) acc[am][tn][r] = 0.0f;

    // stage one 128x64-half tile of A and of BT into LDS buffer p
    auto stage = [&](int p, int k0) {
        #pragma unroll
        for (int c = 0; c < 4; ++c) {
            int idx = (tid + c * 256) * 8;   // half index in tile, 16B chunks
            int row = idx >> 6;              // /64
            int col = idx & 63;
            async_copy_b128(&lds_a[p][idx], &A [(size_t)(bm + row) * K + k0 + col]);
            async_copy_b128(&lds_b[p][idx], &BT[(size_t)(bn + row) * K + k0 + col]);
        }
    };

    int p = 0;
    stage(0, 0);
    wait_async_lds();
    __syncthreads();

    for (int k0 = 0; k0 < K; k0 += 64) {
        if (k0 + 64 < K)
            stage(p ^ 1, k0 + 64);               // async DMA overlaps WMMAs below
        if (k0 + 128 < K) {                       // keep L2 warm one stage further
            __builtin_prefetch(&A [(size_t)(bm + (tid >> 1)) * K + k0 + 128], 0, 2);
            __builtin_prefetch(&BT[(size_t)(bn + (tid >> 1)) * K + k0 + 128], 0, 2);
        }

        #pragma unroll
        for (int ks = 0; ks < 2; ++ks) {
            // two A fragments (rows wr*32 + {0,16}):
            // elems 0..7 -> K = ks*32 + h*8 + 0..7 ; elems 8..15 -> +16
            v16h afrag[2];
            #pragma unroll
            for (int am = 0; am < 2; ++am) {
                const int ra = (wr * 32 + am * 16 + mrow) * 64 + ks * 32 + h * 8;
                v8h alo = *(const v8h*)&lds_a[p][ra];
                v8h ahi = *(const v8h*)&lds_a[p][ra + 16];
                #pragma unroll
                for (int i = 0; i < 8; ++i) { afrag[am][i] = alo[i]; afrag[am][i + 8] = ahi[i]; }
            }
            // four B fragments (cols wc*64 + tn*16), each reused by both A frags
            #pragma unroll
            for (int tn = 0; tn < 4; ++tn) {
                const int rb = (wc * 64 + tn * 16 + mrow) * 64 + ks * 32 + h * 8;
                v8h blo = *(const v8h*)&lds_b[p][rb];
                v8h bhi = *(const v8h*)&lds_b[p][rb + 16];
                v16h bfrag;
                #pragma unroll
                for (int i = 0; i < 8; ++i) { bfrag[i] = blo[i]; bfrag[i + 8] = bhi[i]; }
                acc[0][tn] = __builtin_amdgcn_wmma_f32_16x16x32_f16(
                    false, afrag[0], false, bfrag, (short)0, acc[0][tn], false, false);
                acc[1][tn] = __builtin_amdgcn_wmma_f32_16x16x32_f16(
                    false, afrag[1], false, bfrag, (short)0, acc[1][tn], false, false);
            }
        }

        wait_async_lds();
        __syncthreads();
        p ^= 1;
    }

    // ---- epilogue: C/D layout -> M = r + 8*h, N = lane&15 ----
    const int rbase = bm + wr * 32;
    #pragma unroll
    for (int am = 0; am < 2; ++am) {
        #pragma unroll
        for (int tn = 0; tn < 4; ++tn) {
            const int n  = bn + wc * 64 + tn * 16 + mrow;
            const float bv = (n < NbiasReal) ? bias[n] : 0.0f;
            #pragma unroll
            for (int r = 0; r < 8; ++r) {
                float v = acc[am][tn][r] + bv;
                if (epilogue == 1) {
                    v = (v > 20.0f) ? v : __logf(1.0f + __expf(v));
                    v += 1e-4f;
                }
                C[(size_t)(rbase + am * 16 + h * 8 + r) * N + n] = v;
            }
        }
    }
}

// =====================================================================
// Causal depthwise conv (width 4) + SiLU on x_branch (= xz[:, :DI_]).
// Emits fp32 and f16 copies.
// =====================================================================
__global__ __launch_bounds__(256)
void mamba_conv_silu(const float* __restrict__ xz,
                     const float* __restrict__ cw,
                     const float* __restrict__ cb,
                     float* __restrict__ xc,
                     _Float16* __restrict__ xch)
{
    int idx = blockIdx.x * 256 + threadIdx.x;   // over ROWS_*DI_
    if (idx >= ROWS_ * DI_) return;
    int d = idx % DI_;
    int t = (idx / DI_) % L_;
    int b = idx / (DI_ * L_);

    const float w0 = cw[d * 4 + 0], w1 = cw[d * 4 + 1];
    const float w2 = cw[d * 4 + 2], w3 = cw[d * 4 + 3];
    const size_t rs = 2 * DI_;                  // xz row stride
    const float* base = xz + (size_t)b * L_ * rs + d;

    float acc = cb[d];
    if (t >= 3) acc += w0 * base[(size_t)(t - 3) * rs];
    if (t >= 2) acc += w1 * base[(size_t)(t - 2) * rs];
    if (t >= 1) acc += w2 * base[(size_t)(t - 1) * rs];
    acc += w3 * base[(size_t)t * rs];

    float s = acc / (1.0f + __expf(-acc));      // SiLU
    xc[idx]  = s;
    xch[idx] = (_Float16)s;
}

// =====================================================================
// Pack delta_raw (proj_pad cols 0..47) into zero-padded f16 A (ROWS_ x 64).
// =====================================================================
__global__ __launch_bounds__(256)
void mamba_pack_delta(const float* __restrict__ proj, _Float16* __restrict__ a2)
{
    int idx = blockIdx.x * 256 + threadIdx.x;   // ROWS_*64
    if (idx >= ROWS_ * 64) return;
    int r = idx >> 6, c = idx & 63;
    a2[idx] = (_Float16)((c < DTR_) ? proj[(size_t)r * 128 + c] : 0.0f);
}

// =====================================================================
// Selective scan: thread per (b, d, n); state in register; y reduced
// across the 16 state lanes with shfl_xor (LDS swizzle HW).
// =====================================================================
__global__ __launch_bounds__(256)
void mamba_scan(const float* __restrict__ delta,
                const float* __restrict__ xc,
                const float* __restrict__ proj,     // padded rows of 128
                const float* __restrict__ A_log,
                const float* __restrict__ Dskip,
                float* __restrict__ ys)
{
    int g = blockIdx.x * 256 + threadIdx.x;     // 2*1536*16 = 49152
    const int n = g & 15;
    const int d = (g >> 4) % DI_;
    const int b = g / (16 * DI_);

    const float Adn  = -__expf(A_log[d * NS_ + n]);
    const float Ainv = 1.0f / (Adn + 1e-8f);
    const float dsk  = Dskip[d];

    float state = 0.0f;
    for (int t = 0; t < L_; ++t) {
        const size_t row = (size_t)b * L_ + t;
        const float dv = delta[row * DI_ + d];
        const float u  = xc[row * DI_ + d];
        const float* pr = proj + row * 128;
        const float bs = pr[DTR_ + n];          // b_sel
        const float cs = pr[DTR_ + NS_ + n];    // c_sel

        const float a = __expf(dv * Adn);
        state = a * state + (a - 1.0f) * Ainv * bs * u;

        float part = state * cs;
        #pragma unroll
        for (int off = 8; off; off >>= 1)
            part += __shfl_xor(part, off, 32);
        if (n == 0)
            ys[row * DI_ + d] = part + dsk * u;
    }
}

// =====================================================================
// y = ys * silu(z_branch)  ->  f16 for final GEMM
// =====================================================================
__global__ __launch_bounds__(256)
void mamba_gate(const float* __restrict__ ys,
                const float* __restrict__ xz,
                _Float16* __restrict__ yh)
{
    int idx = blockIdx.x * 256 + threadIdx.x;   // ROWS_*DI_
    if (idx >= ROWS_ * DI_) return;
    int d = idx % DI_;
    size_t row = idx / DI_;
    float z  = xz[row * (2 * DI_) + DI_ + d];
    float sz = z / (1.0f + __expf(-z));
    yh[idx]  = (_Float16)(ys[idx] * sz);
}

// =====================================================================
// Host launcher
// =====================================================================
extern "C" void kernel_launch(void* const* d_in, const int* in_sizes, int n_in,
                              void* d_out, int out_size, void* d_ws, size_t ws_size,
                              hipStream_t stream)
{
    (void)in_sizes; (void)n_in; (void)out_size; (void)ws_size;

    const float* x         = (const float*)d_in[0];
    const float* norm_w    = (const float*)d_in[1];
    const float* norm_b    = (const float*)d_in[2];
    const float* in_proj_w = (const float*)d_in[3];
    const float* in_proj_b = (const float*)d_in[4];
    const float* conv_w    = (const float*)d_in[5];
    const float* conv_b    = (const float*)d_in[6];
    const float* x_proj_w  = (const float*)d_in[7];
    const float* x_proj_b  = (const float*)d_in[8];
    const float* dt_proj_w = (const float*)d_in[9];
    const float* dt_proj_b = (const float*)d_in[10];
    const float* A_log     = (const float*)d_in[11];
    const float* D_skip    = (const float*)d_in[12];
    const float* out_proj_w= (const float*)d_in[13];
    const float* out_proj_b= (const float*)d_in[14];
    float* out             = (float*)d_out;

    // ---- workspace layout ----
    char* wsp = (char*)d_ws;
    auto take = [&](size_t bytes) -> void* {
        void* p = (void*)wsp;
        wsp += (bytes + 255) & ~(size_t)255;
        return p;
    };
    _Float16* xnorm_h = (_Float16*)take((size_t)ROWS_ * D_  * 2);
    _Float16* wt1     = (_Float16*)take((size_t)3072 * 768  * 2);   // in_proj^T
    _Float16* wt2     = (_Float16*)take((size_t)128  * 1536 * 2);   // x_proj^T  (pad N 80->128)
    _Float16* wt3     = (_Float16*)take((size_t)1536 * 64   * 2);   // dt_proj^T (pad K 48->64)
    _Float16* wt4     = (_Float16*)take((size_t)768  * 1536 * 2);   // out_proj^T
    float*    xz      = (float*)   take((size_t)ROWS_ * 3072 * 4);
    float*    xconv   = (float*)   take((size_t)ROWS_ * DI_  * 4);
    _Float16* xconv_h = (_Float16*)take((size_t)ROWS_ * DI_  * 2);
    float*    projp   = (float*)   take((size_t)ROWS_ * 128  * 4);
    _Float16* a2_h    = (_Float16*)take((size_t)ROWS_ * 64   * 2);
    float*    delta   = (float*)   take((size_t)ROWS_ * DI_  * 4);
    float*    ys      = (float*)   take((size_t)ROWS_ * DI_  * 4);
    _Float16* y_h     = (_Float16*)take((size_t)ROWS_ * DI_  * 2);

    const dim3 blk(256);

    // 1) layernorm -> f16
    mamba_layernorm_f16<<<dim3(ROWS_), blk, 0, stream>>>(x, norm_w, norm_b, xnorm_h);

    // 2) weight conversions (transpose + pad + f16)
    mamba_convert_transpose<<<dim3((768  * 3072) / 256), blk, 0, stream>>>(in_proj_w,  wt1, 768,  3072, 768,  3072);
    mamba_convert_transpose<<<dim3((1536 * 128 ) / 256), blk, 0, stream>>>(x_proj_w,   wt2, 1536, 80,   1536, 128);
    mamba_convert_transpose<<<dim3((64   * 1536) / 256), blk, 0, stream>>>(dt_proj_w,  wt3, 48,   1536, 64,   1536);
    mamba_convert_transpose<<<dim3((1536 * 768 ) / 256), blk, 0, stream>>>(out_proj_w, wt4, 1536, 768,  1536, 768);

    // 3) xz = xnorm @ in_proj + b         (M=4096, N=3072, K=768)
    mamba_gemm_wmma<<<dim3(3072 / 128, ROWS_ / 128), blk, 0, stream>>>(
        xnorm_h, wt1, in_proj_b, xz, ROWS_, 3072, 768, 3072, 0);

    // 4) depthwise conv + SiLU
    mamba_conv_silu<<<dim3((ROWS_ * DI_) / 256), blk, 0, stream>>>(xz, conv_w, conv_b, xconv, xconv_h);

    // 5) proj = xconv @ x_proj + b        (M=4096, N=128(pad80), K=1536)
    mamba_gemm_wmma<<<dim3(128 / 128, ROWS_ / 128), blk, 0, stream>>>(
        xconv_h, wt2, x_proj_b, projp, ROWS_, 128, 1536, 80, 0);

    // 6) pack delta_raw -> padded f16
    mamba_pack_delta<<<dim3((ROWS_ * 64) / 256), blk, 0, stream>>>(projp, a2_h);

    // 7) delta = softplus(delta_raw @ dt_proj + b) + 1e-4   (M=4096, N=1536, K=64(pad48))
    mamba_gemm_wmma<<<dim3(1536 / 128, ROWS_ / 128), blk, 0, stream>>>(
        a2_h, wt3, dt_proj_b, delta, ROWS_, 1536, 64, 1536, 1);

    // 8) selective scan
    mamba_scan<<<dim3((B_ * DI_ * NS_) / 256), blk, 0, stream>>>(
        delta, xconv, projp, A_log, D_skip, ys);

    // 9) gate: y = ys * silu(z)
    mamba_gate<<<dim3((ROWS_ * DI_) / 256), blk, 0, stream>>>(ys, xz, y_h);

    // 10) out = y @ out_proj + b          (M=4096, N=768, K=1536)
    mamba_gemm_wmma<<<dim3(768 / 128, ROWS_ / 128), blk, 0, stream>>>(
        y_h, wt4, out_proj_b, out, ROWS_, 768, 1536, 768, 0);
}
